// MultiHeadedAttention_77171972374847
// MI455X (gfx1250) — matrix operand order repaired and checked by reading, not compiled, not run on hardware
//
#include <hip/hip_runtime.h>

// MultiHeadedAttention on MI455X (gfx1250): bf16 WMMA pipeline.
//   k_convW : Wq/Wk/Wv fp32 [128,512] -> bf16 transposed WT [3][512][128]
//   k_proj  : x @ W + b  -> Qs (scaled 1/8, bf16 [B,H,S,64]),
//             Ks (bf16 [B,H,S,64]), VT (relu, bf16 [B,H,64,S])
//   k_attn  : flash attention; K/V tiles staged to LDS with
//             global_load_async_to_lds_b128 (double-buffered, shared by the
//             4 waves of a block), scores/PV via v_wmma_f32_16x16x32_bf16.
// Workspace: WT 384KB + Qs/Ks/VT 3x44MB ~= 133MB.

#define BB  64
#define SS  704
#define HH  8
#define DH  64
#define DIN 128
#define DMODEL 512
#define SBLK 44      // SS/16
#define KBLK 22      // SS/32

typedef __attribute__((ext_vector_type(16))) __bf16 v16bf;
typedef __attribute__((ext_vector_type(8)))  float  v8f;

union Frag {
  v16bf v;
  unsigned int u[8];
};

__device__ __forceinline__ unsigned short f2bf(float f) {
  unsigned int x = __float_as_uint(f);
  return (unsigned short)((x + 0x7FFFu + ((x >> 16) & 1u)) >> 16); // RNE
}
__device__ __forceinline__ unsigned int pk2(float a, float b) {
  return (unsigned int)f2bf(a) | ((unsigned int)f2bf(b) << 16);
}

__device__ __forceinline__ v8f wmma_bf16(const Frag& a, const Frag& b, v8f c) {
  return __builtin_amdgcn_wmma_f32_16x16x32_bf16(
      /*neg_a=*/false, a.v, /*neg_b=*/false, b.v,
      /*c_mod=*/(short)0, c, /*reuse_a=*/false, /*reuse_b=*/false);
}

// Async 16-byte copy global -> LDS (ASYNCcnt-tracked). The low 32 bits of a
// flat shared-aperture pointer are the workgroup-relative LDS byte offset.
__device__ __forceinline__ void async_b128(const void* g, void* lds) {
  unsigned int loff = (unsigned int)(size_t)lds;
  asm volatile("global_load_async_to_lds_b128 %0, %1, off"
               :: "v"(loff), "v"(g) : "memory");
}

// ---------------------------------------------------------------- k_convW
__global__ void k_convW(const float* __restrict__ Wq, const float* __restrict__ Wk,
                        const float* __restrict__ Wv, unsigned short* __restrict__ WT) {
  int idx = blockIdx.x * blockDim.x + threadIdx.x;
  if (idx >= 3 * DMODEL * DIN) return;
  int mat = idx / (DMODEL * DIN);
  int rem = idx - mat * (DMODEL * DIN);
  int n = rem / DIN;
  int k = rem - n * DIN;
  const float* W = (mat == 0) ? Wq : (mat == 1) ? Wk : Wv;
  WT[idx] = f2bf(W[(size_t)k * DMODEL + n]);
}

// ---------------------------------------------------------------- k_proj
__global__ void __launch_bounds__(128) k_proj(
    const float* __restrict__ x, const unsigned short* __restrict__ WT,
    const float* __restrict__ bq, const float* __restrict__ bk,
    const float* __restrict__ bv,
    unsigned short* __restrict__ Qs, unsigned short* __restrict__ Ks,
    unsigned short* __restrict__ VT) {
  const int lane = threadIdx.x & 31;
  const int wid  = blockIdx.x * 4 + (threadIdx.x >> 5);
  const int sblk = wid % SBLK;
  const int h    = (wid / SBLK) % HH;
  const int b    = wid / (SBLK * HH);
  const int s0   = sblk * 16;
  const int half = lane >> 4;
  const int l15  = lane & 15;
  const int bh   = b * HH + h;

  // x A-fragments (fp32 -> bf16) for the 4 K-chunks of DIN=128.
  Frag ax[4];
  {
    const float* xr = x + ((size_t)b * SS + s0 + l15) * DIN;
    for (int c = 0; c < 4; ++c) {
      int k0 = 32 * c + half * 8;
      float4 f0 = *(const float4*)(xr + k0);
      float4 f1 = *(const float4*)(xr + k0 + 4);
      float4 f2 = *(const float4*)(xr + k0 + 16);
      float4 f3 = *(const float4*)(xr + k0 + 20);
      ax[c].u[0] = pk2(f0.x, f0.y); ax[c].u[1] = pk2(f0.z, f0.w);
      ax[c].u[2] = pk2(f1.x, f1.y); ax[c].u[3] = pk2(f1.z, f1.w);
      ax[c].u[4] = pk2(f2.x, f2.y); ax[c].u[5] = pk2(f2.z, f2.w);
      ax[c].u[6] = pk2(f3.x, f3.y); ax[c].u[7] = pk2(f3.z, f3.w);
    }
  }

  v8f acc[3][4];
  for (int m = 0; m < 3; ++m)
    for (int t = 0; t < 4; ++t) acc[m][t] = {};

  for (int c = 0; c < 4; ++c) {
    for (int m = 0; m < 3; ++m) {
      for (int t = 0; t < 4; ++t) {
        int nabs = h * DH + t * 16 + l15;
        const unsigned short* wr =
            WT + ((size_t)m * DMODEL + nabs) * DIN + 32 * c + half * 8;
        Frag bw;
        *(uint4*)&bw.u[0] = *(const uint4*)wr;
        *(uint4*)&bw.u[4] = *(const uint4*)(wr + 16);
        acc[m][t] = wmma_bf16(ax[c], bw, acc[m][t]);
      }
    }
  }

  for (int t = 0; t < 4; ++t) {
    int nloc = t * 16 + l15;
    int nabs = h * DH + nloc;
    float biasq = bq[nabs], biask = bk[nabs], biasv = bv[nabs];
    for (int r = 0; r < 8; ++r) {
      int srow = s0 + r + 8 * half;
      float q  = (acc[0][t][r] + biasq) * 0.125f;   // fold 1/sqrt(64)
      float kk = acc[1][t][r] + biask;
      float vv = acc[2][t][r] + biasv;
      vv = vv > 0.f ? vv : 0.f;                      // ReLU on V
      Qs[((size_t)bh * SS + srow) * DH + nloc] = f2bf(q);
      Ks[((size_t)bh * SS + srow) * DH + nloc] = f2bf(kk);
      VT[((size_t)bh * DH + nloc) * SS + srow] = f2bf(vv);
    }
  }
}

// ---------------------------------------------------------------- k_attn
__global__ void __launch_bounds__(128) k_attn(
    const unsigned short* __restrict__ Qs, const unsigned short* __restrict__ Ks,
    const unsigned short* __restrict__ VT, float* __restrict__ out) {
  // Double-buffered staged tiles: K tile [32 keys][64 d], V tile [64 d][32 keys]
  __shared__ __align__(16) unsigned short sK[2][32 * DH];
  __shared__ __align__(16) unsigned short sV[2][DH * 32];
  __shared__ __align__(16) unsigned short sP[4][16 * 32];  // per-wave P tile

  const int tid  = threadIdx.x;
  const int lane = tid & 31;
  const int wave = tid >> 5;
  const int half = lane >> 4;
  const int l15  = lane & 15;
  const int qb   = blockIdx.x % 11;
  const int h    = (blockIdx.x / 11) % HH;
  const int b    = blockIdx.x / (11 * HH);
  const int bh   = b * HH + h;
  const int s0   = qb * 64 + wave * 16;
  unsigned short* myP = sP[wave];

  const unsigned short* Kbase = Ks + (size_t)bh * SS * DH;
  const unsigned short* Vbase = VT + (size_t)bh * DH * SS;

  // Cooperative async staging: 512 16B chunks per tile pair, 4 per thread.
  auto stage = [&](int k0, int buf) {
    for (int i = 0; i < 2; ++i) {
      int c = tid + 128 * i;  // 0..255
      async_b128(Kbase + (size_t)(k0 + (c >> 3)) * DH + (c & 7) * 8,
                 &sK[buf][c * 8]);
      async_b128(Vbase + (size_t)(c >> 2) * SS + k0 + (c & 3) * 8,
                 &sV[buf][c * 8]);
    }
  };

  // Q A-fragments (d chunks 0..31, 32..63), already scaled by 1/sqrt(Dh).
  Frag aq[2];
  {
    const unsigned short* qr = Qs + ((size_t)bh * SS + s0 + l15) * DH;
    for (int j = 0; j < 2; ++j) {
      const unsigned short* p = qr + 32 * j + half * 8;
      *(uint4*)&aq[j].u[0] = *(const uint4*)p;
      *(uint4*)&aq[j].u[4] = *(const uint4*)(p + 16);
    }
  }

  float mrow[8], lrow[8], alpha[8];
  v8f O[4];
  for (int r = 0; r < 8; ++r) { mrow[r] = -3.0e38f; lrow[r] = 0.f; }
  for (int t = 0; t < 4; ++t) O[t] = {};

  stage(0, 0);

  for (int kb = 0; kb < KBLK; ++kb) {
    const int cur = kb & 1;
    asm volatile("s_wait_asynccnt 0x0" ::: "memory");  // my staged writes done
    __syncthreads();                                   // everyone's writes done;
                                                       // prev reads of buf cur^1 done
    if (kb + 1 < KBLK) stage((kb + 1) * 32, cur ^ 1);

    const unsigned short* sKc = sK[cur];
    const unsigned short* sVc = sV[cur];

    // ---- scores: S[16 q x 32 k] as two 16x16 C tiles (B-frags from LDS)
    v8f Sa = {}, Sb = {};
    for (int j = 0; j < 2; ++j) {
      for (int g = 0; g < 2; ++g) {
        const unsigned short* kr = sKc + (16 * g + l15) * DH + 32 * j + half * 8;
        Frag bk_;
        *(uint4*)&bk_.u[0] = *(const uint4*)kr;
        *(uint4*)&bk_.u[4] = *(const uint4*)(kr + 16);
        if (g == 0) Sa = wmma_bf16(aq[j], bk_, Sa);
        else        Sb = wmma_bf16(aq[j], bk_, Sb);
      }
    }
    // ---- online softmax (row = VGPR index within 16-lane half-group)
    for (int r = 0; r < 8; ++r) {
      float rm = fmaxf(Sa[r], Sb[r]);
      for (int msk = 1; msk < 16; msk <<= 1)
        rm = fmaxf(rm, __shfl_xor(rm, msk, 32));
      float mn = fmaxf(mrow[r], rm);
      alpha[r] = __expf(mrow[r] - mn);
      mrow[r]  = mn;
      float pa = __expf(Sa[r] - mn);
      float pb = __expf(Sb[r] - mn);
      float rs = pa + pb;
      for (int msk = 1; msk < 16; msk <<= 1)
        rs += __shfl_xor(rs, msk, 32);
      lrow[r] = lrow[r] * alpha[r] + rs;
      int m = r + 8 * half;
      myP[m * 32 + l15]      = f2bf(pa);   // C-layout -> row-major LDS
      myP[m * 32 + 16 + l15] = f2bf(pb);
    }
    asm volatile("s_wait_dscnt 0x0" ::: "memory");  // same-wave LDS RAW
    // ---- reload P in A-fragment layout
    Frag ap;
    {
      const unsigned short* p = myP + l15 * 32 + half * 8;
      *(uint4*)&ap.u[0] = *(const uint4*)p;
      *(uint4*)&ap.u[4] = *(const uint4*)(p + 16);
    }
    // ---- rescale O, accumulate P @ V (B-frags from LDS)
    for (int t = 0; t < 4; ++t)
      for (int r = 0; r < 8; ++r) O[t][r] *= alpha[r];
    for (int t = 0; t < 4; ++t) {
      const unsigned short* vr = sVc + (16 * t + l15) * 32 + half * 8;
      Frag bv_;
      *(uint4*)&bv_.u[0] = *(const uint4*)vr;
      *(uint4*)&bv_.u[4] = *(const uint4*)(vr + 16);
      O[t] = wmma_bf16(ap, bv_, O[t]);
    }
  }

  for (int r = 0; r < 8; ++r) lrow[r] = 1.0f / lrow[r];
  for (int t = 0; t < 4; ++t) {
    for (int r = 0; r < 8; ++r) {
      int m = r + 8 * half;
      out[((size_t)b * SS + s0 + m) * DMODEL + h * DH + t * 16 + l15] =
          O[t][r] * lrow[r];
    }
  }
}

// ---------------------------------------------------------------- launch
extern "C" void kernel_launch(void* const* d_in, const int* in_sizes, int n_in,
                              void* d_out, int out_size, void* d_ws, size_t ws_size,
                              hipStream_t stream) {
  (void)in_sizes; (void)n_in; (void)out_size; (void)ws_size;
  const float* x  = (const float*)d_in[0];
  const float* Wq = (const float*)d_in[1];
  const float* bq = (const float*)d_in[2];
  const float* Wk = (const float*)d_in[3];
  const float* bk = (const float*)d_in[4];
  const float* Wv = (const float*)d_in[5];
  const float* bv = (const float*)d_in[6];
  float* out = (float*)d_out;

  char* ws = (char*)d_ws;
  const size_t WT_BYTES  = (size_t)3 * DMODEL * DIN * 2;       // 393216
  const size_t MAT_BYTES = (size_t)BB * HH * SS * DH * 2;      // 46137344
  unsigned short* WT = (unsigned short*)(ws);
  unsigned short* Qs = (unsigned short*)(ws + WT_BYTES);
  unsigned short* Ks = (unsigned short*)(ws + WT_BYTES + MAT_BYTES);
  unsigned short* VT = (unsigned short*)(ws + WT_BYTES + 2 * MAT_BYTES);

  k_convW<<<(3 * DMODEL * DIN + 255) / 256, 256, 0, stream>>>(Wq, Wk, Wv, WT);
  k_proj<<<(BB * HH * SBLK) / 4, 128, 0, stream>>>(x, WT, bq, bk, bv, Qs, Ks, VT);
  k_attn<<<BB * HH * 11, 128, 0, stream>>>(Qs, Ks, VT, out);
}
